// FusionShiftingGraph_36438502539658
// MI455X (gfx1250) — compile-verified, matrix operand order checked by reading, not run on hardware
//
#include <hip/hip_runtime.h>
#include <hip/hip_bf16.h>

typedef __attribute__((ext_vector_type(16))) _Float16 v16h;
typedef __attribute__((ext_vector_type(8)))  _Float16 v8h;
typedef __attribute__((ext_vector_type(8)))  float    v8f;
typedef int b128i __attribute__((vector_size(16)));
typedef __attribute__((address_space(1))) b128i* g128p;   // global
typedef __attribute__((address_space(3))) b128i* l128p;   // LDS

#define Bsz   8
#define NTOK  800
#define Dm    768
#define Hh    8
#define HD    96
#define Mrows (Bsz * NTOK)   // 6400
#define NQKV  2304
#define NEGV  (-1e10f)
#define SQRT_D 27.712812921102035f       // sqrt(768)
#define QSCALE 0.10206207261596575f      // 96^-0.5

#define LDSPITCH 40   // halves per staged row (32 data + 8 pad, keeps 16B align)

#if defined(__has_builtin)
#if __has_builtin(__builtin_amdgcn_global_load_async_to_lds_b128)
#define HAVE_ASYNC 1
#endif
#endif
#ifndef HAVE_ASYNC
#define HAVE_ASYNC 0
#endif

// ---------------------------------------------------------------- helpers
__device__ __forceinline__ v8f wmma16x16x32(v16h a, v16h b, v8f c) {
  return __builtin_amdgcn_wmma_f32_16x16x32_f16(false, a, false, b,
                                                (short)0, c, false, false);
}

// Load one 16-half fragment (A or B operand) from a K-contiguous row.
// Per ISA 16-bit A layout: lane holds K = {kb..kb+7, 16+kb..16+kb+7};
// caller passes p = &row[kBase + (lane>>4)*8].
__device__ __forceinline__ v16h ldfrag(const _Float16* p) {
  v8h c0 = *reinterpret_cast<const v8h*>(p);
  v8h c1 = *reinterpret_cast<const v8h*>(p + 16);
  return __builtin_shufflevector(c0, c1, 0,1,2,3,4,5,6,7,8,9,10,11,12,13,14,15);
}

__device__ __forceinline__ v8f zero8() {
  v8f z = {0.f,0.f,0.f,0.f,0.f,0.f,0.f,0.f};
  return z;
}

__device__ __forceinline__ int mod3(int n) { return (n >= 50) + (n >= 425); }

// 16-byte copy global -> LDS: async (ASYNCcnt-tracked) on CDNA5, sync fallback.
__device__ __forceinline__ void cp16_g2l(const _Float16* g, _Float16* l) {
#if HAVE_ASYNC
  __builtin_amdgcn_global_load_async_to_lds_b128(
      (g128p)(void*)g, (l128p)(void*)l, 0, 0);
#else
  *reinterpret_cast<v8h*>(l) = *reinterpret_cast<const v8h*>(g);
#endif
}

// ---------------------------------------------------------------- embed
__global__ __launch_bounds__(256)
void embed_kernel(const float* __restrict__ s0, const float* __restrict__ s1,
                  const float* __restrict__ s2, _Float16* __restrict__ Xh) {
  int idx = blockIdx.x * 256 + threadIdx.x;
  if (idx >= Mrows * Dm) return;
  int d = idx % Dm;
  int row = idx / Dm;
  int b = row / NTOK, n = row % NTOK;
  float x; int loc;
  if (n < 50)        { x = s0[((size_t)b*50  + n      )*Dm + d]; loc = n;       }
  else if (n < 425)  { x = s1[((size_t)b*375 + (n-50) )*Dm + d]; loc = n - 50;  }
  else               { x = s2[((size_t)b*375 + (n-425))*Dm + d]; loc = n - 425; }
  float pos  = (float)(loc + 2);                       // padding_idx+1 start
  int   hd2  = (d < 384) ? d : d - 384;
  float freq = __expf((float)hd2 * (-9.210340371976184f / 383.0f));
  float ang  = pos * freq;
  float pe   = (d < 384) ? __sinf(ang) : __cosf(ang);
  Xh[idx] = (_Float16)(SQRT_D * x + pe);
}

// ---------------------------------------------------------------- f32 -> f16
__global__ __launch_bounds__(256)
void cvt_f32_f16(const float* __restrict__ in, _Float16* __restrict__ out, int n) {
  int i = blockIdx.x * 256 + threadIdx.x;
  if (i < n) out[i] = (_Float16)in[i];
}

// ---------------------------------------------------------------- GEMM
// C[M x N] = A[M x 768] * W[N x 768]^T + bias ; f16 in, f32 acc.
// Block: 256 thr = 8 waves; block tile 64(M) x 256(N); wave tile 32 x 64.
// A/B K-slabs double-buffered in LDS via async global->LDS copies (ASYNCcnt).
__global__ __launch_bounds__(256)
void gemm_wmma(const _Float16* __restrict__ A, const _Float16* __restrict__ W,
               const float* __restrict__ bias, int N,
               _Float16* __restrict__ Ch, float* __restrict__ Cf, int qkv_mode) {
  __shared__ _Float16 As[2][64 * LDSPITCH];    //  2 x  5 KB
  __shared__ _Float16 Bs[2][256 * LDSPITCH];   //  2 x 20 KB

  const int tid  = threadIdx.x;
  const int lane = tid & 31;
  const int wave = tid >> 5;
  const int lh = lane & 15, hi = lane >> 4;
  const int mBlk = blockIdx.x * 64;
  const int nBlk = blockIdx.y * 256;
  const int mW = (wave >> 2) * 32;         // wave M offset in block tile
  const int nW = (wave & 3) * 64;          // wave N offset in block tile

  v8f acc[2][4];
#pragma unroll
  for (int mt = 0; mt < 2; ++mt)
#pragma unroll
    for (int nt = 0; nt < 4; ++nt) acc[mt][nt] = zero8();

  // 320 rows x 32 halves per K-slab = 1280 16B chunks; 5 per thread.
  auto stage = [&](int k0, int buf) {
#pragma unroll
    for (int i = 0; i < 5; ++i) {
      const int c = tid + i * 256;
      const int row = c >> 2;
      const int sub = (c & 3) * 8;
      if (row < 64) {
        cp16_g2l(A + (size_t)(mBlk + row) * Dm + k0 + sub,
                 &As[buf][row * LDSPITCH + sub]);
      } else {
        const int r2 = row - 64;
        cp16_g2l(W + (size_t)(nBlk + r2) * Dm + k0 + sub,
                 &Bs[buf][r2 * LDSPITCH + sub]);
      }
    }
  };

  stage(0, 0);
  const int KSTEPS = Dm / 32;              // 24
  for (int ki = 0; ki < KSTEPS; ++ki) {
    const int cur = ki & 1;
    if (ki + 1 < KSTEPS) {
      stage((ki + 1) * 32, cur ^ 1);
      // 5 prefetch ops in flight for next buffer; ASYNCcnt retires in order,
      // so <=5 guarantees the current buffer's copies have landed.
      asm volatile("s_wait_asynccnt 5" ::: "memory");
    } else {
      asm volatile("s_wait_asynccnt 0" ::: "memory");
    }
    __syncthreads();                       // all waves' copies visible

    const _Float16* aB = &As[cur][0];
    const _Float16* bB = &Bs[cur][0];
    v16h a0 = ldfrag(aB + (mW + lh)      * LDSPITCH + hi * 8);
    v16h a1 = ldfrag(aB + (mW + 16 + lh) * LDSPITCH + hi * 8);
#pragma unroll
    for (int nt = 0; nt < 4; ++nt) {
      v16h bf = ldfrag(bB + (nW + nt * 16 + lh) * LDSPITCH + hi * 8);
      acc[0][nt] = wmma16x16x32(a0, bf, acc[0][nt]);
      acc[1][nt] = wmma16x16x32(a1, bf, acc[1][nt]);
    }
    __syncthreads();                       // done reading before overwrite
  }

#pragma unroll
  for (int mt = 0; mt < 2; ++mt) {
#pragma unroll
    for (int nt = 0; nt < 4; ++nt) {
      const int col = nBlk + nW + nt * 16 + lh;
      const float bv = bias[col];
      const float sc = (qkv_mode && col < Dm) ? QSCALE : 1.0f;  // q-scale fold
#pragma unroll
      for (int r = 0; r < 8; ++r) {
        const int row = mBlk + mW + mt * 16 + r + 8 * hi;
        const float v = (acc[mt][nt][r] + bv) * sc;
        Ch[(size_t)row * N + col] = (_Float16)v;
        if (Cf) Cf[(size_t)row * N + col] = v;
      }
    }
  }
}

// ---------------------------------------------------------------- attention
// One wave per (b, h, 16-query tile). Flash-style streaming over 32-key tiles.
__global__ __launch_bounds__(32)
void attn_wmma(const _Float16* __restrict__ qkv, _Float16* __restrict__ AO,
               int sameNeg) {
  __shared__ _Float16 Vt[HD][32];   // V^T staging: [head_dim][key]
  __shared__ _Float16 Pl[16][32];   // P tile: C-layout -> A-layout round-trip

  const int lane = threadIdx.x;
  const int lh = lane & 15, hi = lane >> 4;
  const int b = blockIdx.x >> 3, h = blockIdx.x & 7;
  const int q0 = blockIdx.y * 16;
  const _Float16* base = qkv + (size_t)b * NTOK * NQKV;

  // Q fragments (head-dim K = 96 = 3 steps of 32), scale already folded in
  v16h qf[3];
  {
    const _Float16* qrow = base + (size_t)(q0 + lh) * NQKV + h * HD;
#pragma unroll
    for (int kc = 0; kc < 3; ++kc) qf[kc] = ldfrag(qrow + kc * 32 + hi * 8);
  }

  v8f o[6];
#pragma unroll
  for (int nt = 0; nt < 6; ++nt) o[nt] = zero8();
  float mrow[8], lrow[8];
  int qmod[8];
#pragma unroll
  for (int r = 0; r < 8; ++r) {
    mrow[r] = -3.0e38f; lrow[r] = 0.f;
    qmod[r] = mod3(q0 + r + 8 * hi);
  }

  for (int kt = 0; kt < NTOK / 32; ++kt) {
    const int kbase = kt * 32;

    // ---- stage V^T (each lane owns one key row)
    {
      const _Float16* vrow = base + (size_t)(kbase + lane) * NQKV + 2 * Dm + h * HD;
      v8h tmp[12];
#pragma unroll
      for (int t = 0; t < 12; ++t)
        tmp[t] = *reinterpret_cast<const v8h*>(vrow + t * 8);
#pragma unroll
      for (int t = 0; t < 12; ++t)
#pragma unroll
        for (int e = 0; e < 8; ++e) Vt[t * 8 + e][lane] = tmp[t][e];
    }

    // ---- scores: two 16x16 tiles via WMMA (Q x K^T), + analytic block mask
    v8f s[2];
#pragma unroll
    for (int j = 0; j < 2; ++j) {
      s[j] = zero8();
      const _Float16* krow =
          base + (size_t)(kbase + j * 16 + lh) * NQKV + Dm + h * HD;
#pragma unroll
      for (int kc = 0; kc < 3; ++kc) {
        v16h kf = ldfrag(krow + kc * 32 + hi * 8);
        s[j] = wmma16x16x32(qf[kc], kf, s[j]);
      }
      const int kmod = mod3(kbase + j * 16 + lh);
#pragma unroll
      for (int r = 0; r < 8; ++r) {
        const bool neg = (qmod[r] == kmod) == (sameNeg != 0);
        s[j][r] += neg ? NEGV : 0.0f;
      }
    }

    // ---- online softmax (rows live within one 16-lane half)
#pragma unroll
    for (int r = 0; r < 8; ++r) {
      float v = fmaxf(s[0][r], s[1][r]);
#pragma unroll
      for (int off = 8; off; off >>= 1) v = fmaxf(v, __shfl_xor(v, off, 16));
      const float mnew = fmaxf(mrow[r], v);
      const float alpha = __expf(mrow[r] - mnew);
      const float p0 = __expf(s[0][r] - mnew);
      const float p1 = __expf(s[1][r] - mnew);
      s[0][r] = p0; s[1][r] = p1;
      float rs = p0 + p1;
#pragma unroll
      for (int off = 8; off; off >>= 1) rs += __shfl_xor(rs, off, 16);
      lrow[r] = lrow[r] * alpha + rs;
      mrow[r] = mnew;
#pragma unroll
      for (int nt = 0; nt < 6; ++nt) o[nt][r] *= alpha;
    }

    // ---- P: C-layout -> LDS -> A-layout
#pragma unroll
    for (int r = 0; r < 8; ++r) {
      Pl[r + 8 * hi][lh]      = (_Float16)s[0][r];
      Pl[r + 8 * hi][16 + lh] = (_Float16)s[1][r];
    }
    asm volatile("s_wait_dscnt 0" ::: "memory");   // CDNA5 split DS counter
    __builtin_amdgcn_wave_barrier();

    v16h pf = ldfrag(&Pl[lh][hi * 8]);

    // ---- O += P(16x32) x V(32x96): 6 WMMAs, B frags contiguous in Vt rows
#pragma unroll
    for (int nt = 0; nt < 6; ++nt) {
      v16h vf = ldfrag(&Vt[nt * 16 + lh][hi * 8]);
      o[nt] = wmma16x16x32(pf, vf, o[nt]);
    }
    __builtin_amdgcn_wave_barrier();               // keep LDS reuse ordered
  }

  // ---- finalize: divide by row sums, write f16
  float inv[8];
#pragma unroll
  for (int r = 0; r < 8; ++r) inv[r] = 1.0f / lrow[r];
#pragma unroll
  for (int nt = 0; nt < 6; ++nt)
#pragma unroll
    for (int r = 0; r < 8; ++r) {
      const int row = b * NTOK + q0 + r + 8 * hi;
      AO[(size_t)row * Dm + h * HD + nt * 16 + lh] = (_Float16)(o[nt][r] * inv[r]);
    }
}

// ---------------------------------------------------------------- layernorm
__global__ __launch_bounds__(256)
void ln_kernel(const float* __restrict__ X, const float* __restrict__ g,
               const float* __restrict__ bta, float* __restrict__ out) {
  __shared__ float red[256];
  const int t = threadIdx.x;
  const size_t row = blockIdx.x;
  const float* x = X + row * Dm;
  const float v0 = x[t], v1 = x[t + 256], v2 = x[t + 512];
  red[t] = v0 + v1 + v2;
  __syncthreads();
  for (int off = 128; off; off >>= 1) {
    if (t < off) red[t] += red[t + off];
    __syncthreads();
  }
  const float mu = red[0] * (1.0f / Dm);
  __syncthreads();
  const float d0 = v0 - mu, d1 = v1 - mu, d2 = v2 - mu;
  red[t] = d0 * d0 + d1 * d1 + d2 * d2;
  __syncthreads();
  for (int off = 128; off; off >>= 1) {
    if (t < off) red[t] += red[t + off];
    __syncthreads();
  }
  const float rstd = rsqrtf(red[0] * (1.0f / Dm) + 1e-5f);
  float* o = out + row * Dm;
  o[t]       = d0 * rstd * g[t]       + bta[t];
  o[t + 256] = d1 * rstd * g[t + 256] + bta[t + 256];
  o[t + 512] = d2 * rstd * g[t + 512] + bta[t + 512];
}

// ---------------------------------------------------------------- host
extern "C" void kernel_launch(void* const* d_in, const int* in_sizes, int n_in,
                              void* d_out, int out_size, void* d_ws, size_t ws_size,
                              hipStream_t stream) {
  const float* s0 = (const float*)d_in[0];
  const float* s1 = (const float*)d_in[1];
  const float* s2 = (const float*)d_in[2];

  // workspace carve (halves first, floats after — all 16B aligned)
  _Float16* Xh    = (_Float16*)d_ws;                       // 6400*768
  _Float16* QKV   = Xh    + (size_t)Mrows * Dm;            // 6400*2304
  _Float16* AO    = QKV   + (size_t)Mrows * NQKV;          // 6400*768
  _Float16* Whin  = AO    + (size_t)Mrows * Dm;            // 2304*768
  _Float16* Whout = Whin  + (size_t)NQKV * Dm;             // 768*768
  float*    Xf    = (float*)(Whout + (size_t)Dm * Dm);     // 6400*768 f32

  for (int g = 0; g < 2; ++g) {
    const int ib = 3 + g * 6;
    const float* w_in  = (const float*)d_in[ib + 0];
    const float* b_in  = (const float*)d_in[ib + 1];
    const float* w_out = (const float*)d_in[ib + 2];
    const float* b_out = (const float*)d_in[ib + 3];
    const float* ln_g  = (const float*)d_in[ib + 4];
    const float* ln_b  = (const float*)d_in[ib + 5];

    embed_kernel<<<(Mrows * Dm + 255) / 256, 256, 0, stream>>>(s0, s1, s2, Xh);

    for (int l = 0; l < 4; ++l) {
      cvt_f32_f16<<<(NQKV * Dm + 255) / 256, 256, 0, stream>>>(
          w_in + (size_t)l * NQKV * Dm, Whin, NQKV * Dm);
      cvt_f32_f16<<<(Dm * Dm + 255) / 256, 256, 0, stream>>>(
          w_out + (size_t)l * Dm * Dm, Whout, Dm * Dm);

      gemm_wmma<<<dim3(Mrows / 64, NQKV / 256), 256, 0, stream>>>(
          Xh, Whin, b_in + (size_t)l * NQKV, NQKV, QKV, nullptr, 1);

      attn_wmma<<<dim3(Bsz * Hh, NTOK / 16), 32, 0, stream>>>(
          QKV, AO, g == 0 ? 1 : 0);

      gemm_wmma<<<dim3(Mrows / 64, Dm / 256), 256, 0, stream>>>(
          AO, Whout, b_out + (size_t)l * Dm, Dm, Xh, Xf, 0);
    }

    ln_kernel<<<Mrows, 256, 0, stream>>>(Xf, ln_g, ln_b,
                                         (float*)d_out + (size_t)g * Mrows * Dm);
  }
}
